// BolmoLocalLayer_52183852646975
// MI455X (gfx1250) — matrix operand order repaired and checked by reading, not compiled, not run on hardware
//
#include <hip/hip_runtime.h>

// ---------------- types ----------------
typedef __bf16 bf16_t;
typedef __attribute__((ext_vector_type(2)))  __bf16 bf2;
typedef __attribute__((ext_vector_type(4)))  __bf16 bf4;
typedef __attribute__((ext_vector_type(8)))  __bf16 bf8;
typedef __attribute__((ext_vector_type(16))) __bf16 v16bf;
typedef __attribute__((ext_vector_type(8)))  float  v8f;
typedef __attribute__((ext_vector_type(4)))  float  f4v;
typedef __attribute__((ext_vector_type(4)))  int    v4i;

static constexpr int kB  = 2;
static constexpr int kS  = 1024;
static constexpr int kD  = 2048;
static constexpr int kNH = 8;
static constexpr int kQK = 1024;   // NH * 128
static constexpr int kV  = 2048;   // NH * 256
static constexpr int kI  = 8192;
static constexpr int kDQK = 128;
static constexpr int kDV  = 256;
static constexpr float kEPS = 1e-6f;

// ---------------- explicit global-address-space loads ----------------
// Forces global_load_b128 (not flat_load_b128, which also ties up DScnt/LDS
// pipe) for staging reads whose provenance the compiler can't infer.
__device__ __forceinline__ f4v ld_g_f4v(const float* p) {
  return *(const __attribute__((address_space(1))) f4v*)p;
}
__device__ __forceinline__ bf8 ld_g_bf8(const bf16_t* p) {
  return *(const __attribute__((address_space(1))) bf8*)p;
}

// ---------------- gfx1250 async global->LDS copy (guarded) ----------------
#if defined(__has_builtin)
#if __has_builtin(__builtin_amdgcn_global_load_async_to_lds_b128)
#define HAS_ASYNC_LDS 1
#endif
#if __has_builtin(__builtin_amdgcn_s_wait_asynccnt)
#define HAS_WAIT_ASYNC 1
#endif
#endif

// copy 16 bytes (8 bf16) global -> LDS (16B aligned on both sides)
__device__ __forceinline__ void copy16_g2l(const bf16_t* g, bf16_t* l) {
#ifdef HAS_ASYNC_LDS
  v4i* gg = (v4i*)g;  // C-style cast drops const
  v4i* ll = (v4i*)l;
  __builtin_amdgcn_global_load_async_to_lds_b128(
      (__attribute__((address_space(1))) v4i*)gg,
      (__attribute__((address_space(3))) v4i*)ll, 0, 0);
#else
  *(bf8*)l = ld_g_bf8(g);
#endif
}
__device__ __forceinline__ void copy_join() {
#ifdef HAS_ASYNC_LDS
#ifdef HAS_WAIT_ASYNC
  __builtin_amdgcn_s_wait_asynccnt(0);
#else
  asm volatile("s_wait_asynccnt 0x0" ::: "memory");
#endif
#endif
}

// ---------------- WMMA helpers ----------------
__device__ __forceinline__ v8f wmma_bf16(v16bf a, v16bf b, v8f c) {
  return __builtin_amdgcn_wmma_f32_16x16x32_bf16(false, a, false, b, (short)0, c,
                                                 false, false);
}

// A-matrix 16x32 bf16: lane = row (lane&15); lanes 0-15 hold K 0..7,16..23;
// lanes 16-31 hold K 8..15,24..31.
__device__ __forceinline__ v16bf frag_a(const bf16_t* rowp, int khalf) {
  bf8 lo = *(const bf8*)(rowp + 8 * khalf);
  bf8 hi = *(const bf8*)(rowp + 16 + 8 * khalf);
  return __builtin_shufflevector(lo, hi, 0, 1, 2, 3, 4, 5, 6, 7, 8, 9, 10, 11,
                                 12, 13, 14, 15);
}
// B-matrix 32x16 bf16 staged transposed in LDS ([n][k]) -> same K order as A.
__device__ __forceinline__ v16bf frag_b(const bf16_t* rowp, int khalf) {
  bf8 lo = *(const bf8*)(rowp + 8 * khalf);
  bf8 hi = *(const bf8*)(rowp + 16 + 8 * khalf);
  return __builtin_shufflevector(lo, hi, 0, 1, 2, 3, 4, 5, 6, 7, 8, 9, 10, 11,
                                 12, 13, 14, 15);
}

__device__ __forceinline__ float wave_sum(float v) {
#pragma unroll
  for (int o = 16; o > 0; o >>= 1) v += __shfl_xor(v, o, 32);
  return v;
}
__device__ __forceinline__ float logsig(float x) {
  return x < 0.f ? x - log1pf(expf(x)) : -log1pf(expf(-x));
}

// ---------------- elementwise convert ----------------
__global__ void f32_to_bf16_kernel(const float* __restrict__ in,
                                   bf16_t* __restrict__ out, int n4) {
  int i = blockIdx.x * blockDim.x + threadIdx.x;
  if (i < n4) {
    f4v v = ((const f4v*)in)[i];
    bf4 o;
    o[0] = (bf16_t)v[0]; o[1] = (bf16_t)v[1];
    o[2] = (bf16_t)v[2]; o[3] = (bf16_t)v[3];
    ((bf4*)out)[i] = o;
  }
}

// ---------------- generic WMMA GEMM (K-step 64) ----------------
// C[M,N] = A[M,K](bf16) @ W[K,N](f32 -> bf16 during LDS staging)
// MODE 0: store f32   MODE 1: store bf16   MODE 2: dual W, store
// bf16( silu(A@W0) * (A@W1) )
template <int MODE>
__global__ __launch_bounds__(256) void gemm_wmma_kernel(
    const bf16_t* __restrict__ A, const float* __restrict__ W0,
    const float* __restrict__ W1, float* __restrict__ Cf,
    bf16_t* __restrict__ Cb, int M, int N, int K) {
  constexpr int ALD = 72;  // 64 + 8 pad; 144B row stride (16B aligned)
  constexpr int BLD = 72;
  __shared__ bf16_t As[128 * ALD];
  __shared__ bf16_t Bs[128 * BLD];
  __shared__ bf16_t Bs2[(MODE == 2) ? 128 * BLD : 1];

  const int tid = threadIdx.x, lane = tid & 31, w = tid >> 5;
  const int wm = w >> 2, wn = w & 3;  // 2 x 4 wave grid
  const int m0 = blockIdx.y * 128, n0 = blockIdx.x * 128;
  const int khalf = lane >> 4, l15 = lane & 15;

  // persistent staging pointers: A tile = 128x64 bf16 = 1024 16B chunks
  const bf16_t* ap[4];
  int aoff[4];
#pragma unroll
  for (int i = 0; i < 4; i++) {
    int g = tid + 256 * i;
    int r = g >> 3, c8 = (g & 7) * 8;
    ap[i] = A + (size_t)(m0 + r) * K + c8;
    aoff[i] = r * ALD + c8;
  }
  // W tile = 64x128 f32; pack row-pairs -> 32-bit bf16x2 LDS stores
  const float* wp[4];
  const float* wp2[4];
  int boff[4];
#pragma unroll
  for (int i = 0; i < 4; i++) {
    int g = tid + 256 * i;
    int kp = g >> 5, c4 = (g & 31) * 4;  // rows 2kp,2kp+1 ; cols c4..c4+3
    wp[i] = W0 + (size_t)(2 * kp) * N + n0 + c4;
    if (MODE == 2) wp2[i] = W1 + (size_t)(2 * kp) * N + n0 + c4;
    boff[i] = c4 * BLD + 2 * kp;
  }

  v8f acc[4][2], acc2[4][2];
#pragma unroll
  for (int a = 0; a < 4; a++)
#pragma unroll
    for (int b = 0; b < 2; b++) {
      acc[a][b] = v8f{0.f, 0.f, 0.f, 0.f, 0.f, 0.f, 0.f, 0.f};
      acc2[a][b] = v8f{0.f, 0.f, 0.f, 0.f, 0.f, 0.f, 0.f, 0.f};
    }

  for (int k0 = 0; k0 < K; k0 += 64) {
#pragma unroll
    for (int i = 0; i < 4; i++) {
      copy16_g2l(ap[i], &As[aoff[i]]);
      ap[i] += 64;
    }
#pragma unroll
    for (int i = 0; i < 4; i++) {
      f4v r0 = ld_g_f4v(wp[i]);
      f4v r1 = ld_g_f4v(wp[i] + N);
#pragma unroll
      for (int j = 0; j < 4; j++) {
        bf2 p;
        p[0] = (bf16_t)r0[j];
        p[1] = (bf16_t)r1[j];
        *(bf2*)&Bs[boff[i] + j * BLD] = p;
      }
      if (MODE == 2) {
        f4v u0 = ld_g_f4v(wp2[i]);
        f4v u1 = ld_g_f4v(wp2[i] + N);
#pragma unroll
        for (int j = 0; j < 4; j++) {
          bf2 p;
          p[0] = (bf16_t)u0[j];
          p[1] = (bf16_t)u1[j];
          *(bf2*)&Bs2[boff[i] + j * BLD] = p;
        }
      }
      if (k0 + 64 < K) {
        __builtin_prefetch(wp[i] + (size_t)64 * N, 0, 1);
        if (MODE == 2) __builtin_prefetch(wp2[i] + (size_t)64 * N, 0, 1);
      }
      wp[i] += (size_t)64 * N;
      if (MODE == 2) wp2[i] += (size_t)64 * N;
    }
    copy_join();
    __syncthreads();

#pragma unroll
    for (int kk = 0; kk < 64; kk += 32) {
      v16bf af[4];
#pragma unroll
      for (int mt = 0; mt < 4; mt++)
        af[mt] = frag_a(&As[(wm * 64 + mt * 16 + l15) * ALD + kk], khalf);
#pragma unroll
      for (int nt = 0; nt < 2; nt++) {
        v16bf bfr = frag_b(&Bs[(wn * 32 + nt * 16 + l15) * BLD + kk], khalf);
#pragma unroll
        for (int mt = 0; mt < 4; mt++)
          acc[mt][nt] = wmma_bf16(af[mt], bfr, acc[mt][nt]);
        if (MODE == 2) {
          v16bf b2 = frag_b(&Bs2[(wn * 32 + nt * 16 + l15) * BLD + kk], khalf);
#pragma unroll
          for (int mt = 0; mt < 4; mt++)
            acc2[mt][nt] = wmma_bf16(af[mt], b2, acc2[mt][nt]);
        }
      }
    }
    __syncthreads();
  }

  // epilogue: C layout -> row = base + r + 8*khalf, col = base + (lane&15)
#pragma unroll
  for (int mt = 0; mt < 4; mt++)
#pragma unroll
    for (int nt = 0; nt < 2; nt++)
#pragma unroll
      for (int r = 0; r < 8; r++) {
        int row = m0 + wm * 64 + mt * 16 + r + 8 * khalf;
        int col = n0 + wn * 32 + nt * 16 + l15;
        float v = acc[mt][nt][r];
        if (MODE == 0) {
          Cf[(size_t)row * N + col] = v;
        } else if (MODE == 1) {
          Cb[(size_t)row * N + col] = (bf16_t)v;
        } else {
          float u = acc2[mt][nt][r];
          float s = v / (1.f + expf(-v));  // silu
          Cb[(size_t)row * N + col] = (bf16_t)(s * u);
        }
      }
}

// ---------------- gate pre-activations ----------------
__global__ __launch_bounds__(256) void gates_kernel(
    const float* __restrict__ x, const float* __restrict__ Wi,
    const float* __restrict__ bi, const float* __restrict__ Wf,
    const float* __restrict__ bfv, float* __restrict__ icap,
    float* __restrict__ fcap) {
  int tok = blockIdx.x;
  int lane = threadIdx.x & 31, h = threadIdx.x >> 5;  // wave per head
  const float* xr = x + (size_t)tok * kD;
  float di = 0.f, df = 0.f;
  for (int d = lane; d < kD; d += 32) {
    float xv = xr[d];
    di += xv * Wi[d * kNH + h];
    df += xv * Wf[d * kNH + h];
  }
  di = wave_sum(di);
  df = wave_sum(df);
  if (lane == 0) {
    int b = tok / kS, t = tok % kS;
    icap[(b * kNH + h) * kS + t] = 15.f * tanhf((di + bi[h]) / 15.f);
    fcap[(b * kNH + h) * kS + t] = 15.f * tanhf((df + bfv[h]) / 15.f);
  }
}

// ---------------- per-(b,h) scan: lfc, a_s, running max ----------------
__global__ void scan_kernel(const float* __restrict__ icap,
                            const float* __restrict__ fcap,
                            float* __restrict__ lfc, float* __restrict__ av,
                            float* __restrict__ Mst) {
  int bh = threadIdx.x;
  if (bh >= kB * kNH) return;
  float lf = 0.f, mv = -1e30f;
  for (int t = 0; t < kS; t++) {
    lf += logsig(fcap[bh * kS + t]);
    lfc[bh * kS + t] = lf;
    float at = icap[bh * kS + t] - lf;
    mv = fmaxf(mv, at);
    av[bh * kS + t] = at;
    Mst[bh * kS + t] = mv;
  }
}

// ---------------- mLSTM stabilized attention (flash-style) ----------------
__global__ __launch_bounds__(256) void attn_kernel(
    const bf16_t* __restrict__ qb, const bf16_t* __restrict__ kb,
    const bf16_t* __restrict__ vb, const float* __restrict__ av,
    const float* __restrict__ Mst, const float* __restrict__ lfc,
    float* __restrict__ hout) {
  constexpr int QLD = 136, KLD = 136, SLD = 40, VLD = 40;
  __shared__ bf16_t Qsm[64 * QLD];   // q tile [t][d]
  __shared__ bf16_t Ksm[32 * KLD];   // k tile [s][d]
  __shared__ bf16_t Ssm[64 * SLD];   // score tile [t][s]
  __shared__ bf16_t Vsm[256 * VLD];  // v tile transposed [n][s]
  __shared__ float rs[64];

  const int tid = threadIdx.x, lane = tid & 31, w = tid >> 5;
  const int khalf = lane >> 4, l15 = lane & 15;
  const int bh = blockIdx.y, b = bh >> 3, h = bh & 7;
  const int jt = blockIdx.x, t0 = jt * 64;
  const float scale = 0.08838834764831845f;  // 128^-0.5

  // async-load Q tile 64x128 (1024 16B chunks)
#pragma unroll
  for (int i = 0; i < 4; i++) {
    int g = tid + 256 * i;
    int r = g >> 4, c8 = (g & 15) * 8;
    copy16_g2l(&qb[(size_t)(b * kS + t0 + r) * kQK + h * kDQK + c8],
               &Qsm[r * QLD + c8]);
  }
  if (tid < 64) rs[tid] = 0.f;

  // persistent staging pointers for K / V tiles
  const bf16_t* kp_[2];
  int koff[2];
#pragma unroll
  for (int i = 0; i < 2; i++) {
    int g = tid + 256 * i;
    int r = g >> 4, c8 = (g & 15) * 8;
    kp_[i] = kb + (size_t)(b * kS + r) * kQK + h * kDQK + c8;
    koff[i] = r * KLD + c8;
  }
  const bf16_t* vp_[2];
  int vn8[2], vsp[2];
#pragma unroll
  for (int i = 0; i < 2; i++) {
    int g = tid + 256 * i;
    int sp = g >> 5, n8 = (g & 31) * 8;  // s rows 2sp,2sp+1 ; cols n8..n8+7
    vp_[i] = vb + (size_t)(b * kS + 2 * sp) * kV + h * kDV + n8;
    vn8[i] = n8;
    vsp[i] = sp;
  }

  v8f o[2][4];
#pragma unroll
  for (int a = 0; a < 2; a++)
#pragma unroll
    for (int c = 0; c < 4; c++)
      o[a][c] = v8f{0.f, 0.f, 0.f, 0.f, 0.f, 0.f, 0.f, 0.f};

  const int nsb = 2 * jt + 2;  // causal: s up to t0+63
  for (int js = 0; js < nsb; js++) {
    const int s0 = js * 32;
    __syncthreads();
    // K tile 32x128 async
#pragma unroll
    for (int i = 0; i < 2; i++) {
      copy16_g2l(kp_[i], &Ksm[koff[i]]);
      kp_[i] += (size_t)32 * kQK;
    }
    // V tile 32x256, transpose with packed bf16x2 stores
#pragma unroll
    for (int i = 0; i < 2; i++) {
      bf8 v0 = ld_g_bf8(vp_[i]);
      bf8 v1 = ld_g_bf8(vp_[i] + kV);
#pragma unroll
      for (int j = 0; j < 8; j++) {
        bf2 p;
        p[0] = v0[j];
        p[1] = v1[j];
        *(bf2*)&Vsm[(vn8[i] + j) * VLD + 2 * vsp[i]] = p;
      }
      vp_[i] += (size_t)32 * kV;
    }
    copy_join();
    __syncthreads();

    // scores: waves 4(t) x 2(s), one 16x16 tile each
    {
      const int tw = w >> 1, sw = w & 1;
      v8f sc = v8f{0.f, 0.f, 0.f, 0.f, 0.f, 0.f, 0.f, 0.f};
#pragma unroll
      for (int d0 = 0; d0 < 128; d0 += 32) {
        v16bf aq = frag_a(&Qsm[(16 * tw + l15) * QLD + d0], khalf);
        v16bf bk = frag_b(&Ksm[(16 * sw + l15) * KLD + d0], khalf);
        sc = wmma_bf16(aq, bk, sc);
      }
      const int sl = 16 * sw + l15;
      const int sg = s0 + sl;
      const float a_s = av[bh * kS + sg];
#pragma unroll
      for (int r = 0; r < 8; r++) {
        int tl = 16 * tw + r + 8 * khalf;
        int tg = t0 + tl;
        float val = (float)sc[r] * scale;
        val = (sg <= tg) ? val * expf(a_s - Mst[bh * kS + tg]) : 0.f;
        Ssm[tl * SLD + sl] = (bf16_t)val;
      }
    }
    __syncthreads();

    if (tid < 64) {
      float sm = 0.f;
#pragma unroll
      for (int s = 0; s < 32; s++) sm += (float)Ssm[tid * SLD + s];
      rs[tid] += sm;
    }

    // O += scores @ V : waves 2(t) x 4(n)
    {
      const int tw2 = w >> 2, nw = w & 3;
#pragma unroll
      for (int mt = 0; mt < 2; mt++) {
        v16bf as_ = frag_a(&Ssm[(32 * tw2 + 16 * mt + l15) * SLD], khalf);
#pragma unroll
        for (int nt = 0; nt < 4; nt++) {
          v16bf bv = frag_b(&Vsm[(64 * nw + 16 * nt + l15) * VLD], khalf);
          o[mt][nt] = wmma_bf16(as_, bv, o[mt][nt]);
        }
      }
    }
  }
  __syncthreads();

  const int tw2 = w >> 2, nw = w & 3;
#pragma unroll
  for (int mt = 0; mt < 2; mt++)
#pragma unroll
    for (int nt = 0; nt < 4; nt++)
#pragma unroll
      for (int r = 0; r < 8; r++) {
        int tl = 32 * tw2 + 16 * mt + r + 8 * khalf;
        int tg = t0 + tl;
        int n = 64 * nw + 16 * nt + l15;
        float m = lfc[bh * kS + tg] + Mst[bh * kS + tg];
        float nrm = fmaxf(fabsf(rs[tl]), expf(-m));
        hout[(size_t)(b * kS + tg) * kV + h * kDV + n] =
            o[mt][nt][r] / (nrm + kEPS);
      }
}

// ---------------- per-head RMS norm * sigmoid(o_preact) ----------------
__global__ __launch_bounds__(256) void gatenorm_kernel(
    const float* __restrict__ hattn, const float* __restrict__ opre,
    const float* __restrict__ mhw, bf16_t* __restrict__ hg) {
  int tok = blockIdx.x;
  int lane = threadIdx.x & 31, h = threadIdx.x >> 5;
  const float* hr = hattn + (size_t)tok * kV + h * kDV;
  float vals[8], ss = 0.f;
#pragma unroll
  for (int j = 0; j < 8; j++) {
    vals[j] = hr[lane + 32 * j];
    ss += vals[j] * vals[j];
  }
  ss = wave_sum(ss);
  float inv = rsqrtf(ss / (float)kDV + kEPS);
#pragma unroll
  for (int j = 0; j < 8; j++) {
    int c = h * kDV + lane + 32 * j;
    float hn = vals[j] * inv * mhw[c];
    float g = 1.f / (1.f + expf(-opre[(size_t)tok * kV + c]));
    hg[(size_t)tok * kV + c] = (bf16_t)(g * hn);
  }
}

// ---------------- out = resid + w * rms_norm(xin); optional bf16 copy -------
__global__ __launch_bounds__(256) void residrms_kernel(
    const float* __restrict__ resid, const float* __restrict__ xin,
    const float* __restrict__ wn, float* __restrict__ outf,
    bf16_t* __restrict__ outb) {
  int tok = blockIdx.x, tid = threadIdx.x, lane = tid & 31, w = tid >> 5;
  __shared__ float red[8];
  const float* xr = xin + (size_t)tok * kD;
  float ss = 0.f;
  for (int d = tid; d < kD; d += 256) {
    float v = xr[d];
    ss += v * v;
  }
  ss = wave_sum(ss);
  if (lane == 0) red[w] = ss;
  __syncthreads();
  if (w == 0) {
    float t = (lane < 8) ? red[lane] : 0.f;
    t = wave_sum(t);
    if (lane == 0) red[0] = t;
  }
  __syncthreads();
  float inv = rsqrtf(red[0] / (float)kD + kEPS);
  for (int d = tid; d < kD; d += 256) {
    float v = resid[(size_t)tok * kD + d] + wn[d] * xr[d] * inv;
    outf[(size_t)tok * kD + d] = v;
    if (outb) outb[(size_t)tok * kD + d] = (bf16_t)v;
  }
}

// ---------------- launch ----------------
extern "C" void kernel_launch(void* const* d_in, const int* /*in_sizes*/,
                              int /*n_in*/, void* d_out, int /*out_size*/,
                              void* d_ws, size_t /*ws_size*/,
                              hipStream_t stream) {
  const float* x    = (const float*)d_in[0];
  const float* Wq   = (const float*)d_in[1];
  const float* Wk   = (const float*)d_in[2];
  const float* Wv   = (const float*)d_in[3];
  const float* Wo   = (const float*)d_in[4];
  const float* Wi   = (const float*)d_in[5];
  const float* bi   = (const float*)d_in[6];
  const float* Wf   = (const float*)d_in[7];
  const float* bfv  = (const float*)d_in[8];
  const float* mhw  = (const float*)d_in[9];
  const float* Wout = (const float*)d_in[10];
  const float* r1w  = (const float*)d_in[11];
  const float* Wg   = (const float*)d_in[12];
  const float* Wu   = (const float*)d_in[13];
  const float* Wd   = (const float*)d_in[14];
  const float* r2w  = (const float*)d_in[15];
  float* out = (float*)d_out;

  char* ws = (char*)d_ws;
  size_t off = 0;
  auto alloc = [&](size_t bytes) {
    size_t o = off;
    off += (bytes + 255) & ~(size_t)255;
    return o;
  };
  const size_t M = (size_t)kB * kS;  // 2048 tokens
  bf16_t* x_bf   = (bf16_t*)(ws + alloc(M * kD * 2));
  bf16_t* q_bf   = (bf16_t*)(ws + alloc(M * kQK * 2));
  bf16_t* k_bf   = (bf16_t*)(ws + alloc(M * kQK * 2));
  bf16_t* v_bf   = (bf16_t*)(ws + alloc(M * kV * 2));
  float*  o_pre  = (float*)(ws + alloc(M * kV * 4));
  float*  icap   = (float*)(ws + alloc((size_t)kB * kNH * kS * 4));
  float*  fcap   = (float*)(ws + alloc((size_t)kB * kNH * kS * 4));
  float*  lfc    = (float*)(ws + alloc((size_t)kB * kNH * kS * 4));
  float*  av     = (float*)(ws + alloc((size_t)kB * kNH * kS * 4));
  float*  Mst    = (float*)(ws + alloc((size_t)kB * kNH * kS * 4));
  float*  h_attn = (float*)(ws + alloc(M * kV * 4));
  bf16_t* h_gate = (bf16_t*)(ws + alloc(M * kV * 2));
  float*  xl_out = (float*)(ws + alloc(M * kD * 4));
  float*  h1     = (float*)(ws + alloc(M * kD * 4));
  bf16_t* h1_bf  = (bf16_t*)(ws + alloc(M * kD * 2));
  bf16_t* act    = (bf16_t*)(ws + alloc(M * kI * 2));
  float*  mlpout = (float*)(ws + alloc(M * kD * 4));

  {
    int n4 = (int)(M * kD / 4);
    f32_to_bf16_kernel<<<(n4 + 255) / 256, 256, 0, stream>>>(x, x_bf, n4);
  }
  gemm_wmma_kernel<1><<<dim3(kQK / 128, M / 128), 256, 0, stream>>>(
      x_bf, Wq, nullptr, nullptr, q_bf, (int)M, kQK, kD);
  gemm_wmma_kernel<1><<<dim3(kQK / 128, M / 128), 256, 0, stream>>>(
      x_bf, Wk, nullptr, nullptr, k_bf, (int)M, kQK, kD);
  gemm_wmma_kernel<1><<<dim3(kV / 128, M / 128), 256, 0, stream>>>(
      x_bf, Wv, nullptr, nullptr, v_bf, (int)M, kV, kD);
  gemm_wmma_kernel<0><<<dim3(kV / 128, M / 128), 256, 0, stream>>>(
      x_bf, Wo, nullptr, o_pre, nullptr, (int)M, kV, kD);
  gates_kernel<<<(int)M, 256, 0, stream>>>(x, Wi, bi, Wf, bfv, icap, fcap);
  scan_kernel<<<1, kB * kNH, 0, stream>>>(icap, fcap, lfc, av, Mst);
  attn_kernel<<<dim3(kS / 64, kB * kNH), 256, 0, stream>>>(q_bf, k_bf, v_bf, av,
                                                           Mst, lfc, h_attn);
  gatenorm_kernel<<<(int)M, 256, 0, stream>>>(h_attn, o_pre, mhw, h_gate);
  gemm_wmma_kernel<0><<<dim3(kD / 128, M / 128), 256, 0, stream>>>(
      h_gate, Wout, nullptr, xl_out, nullptr, (int)M, kD, kV);
  residrms_kernel<<<(int)M, 256, 0, stream>>>(x, xl_out, r1w, h1, h1_bf);
  gemm_wmma_kernel<2><<<dim3(kI / 128, M / 128), 256, 0, stream>>>(
      h1_bf, Wg, Wu, nullptr, act, (int)M, kI, kD);
  gemm_wmma_kernel<0><<<dim3(kD / 128, M / 128), 256, 0, stream>>>(
      act, Wd, nullptr, mlpout, nullptr, (int)M, kD, kI);
  residrms_kernel<<<(int)M, 256, 0, stream>>>(h1, mlpout, r2w, out, nullptr);
}
